// AnomalyDetector_61375082660111
// MI455X (gfx1250) — matrix operand order repaired and checked by reading, not compiled, not run on hardware
//
#include <hip/hip_runtime.h>

typedef __bf16 bf16;
typedef __attribute__((ext_vector_type(8)))  bf16  v8bf;
typedef __attribute__((ext_vector_type(16))) bf16  v16bf;
typedef __attribute__((ext_vector_type(8)))  float v8f;

#define B_   128
#define T_   512
#define D_   16
#define H_   512
#define G4H  2048

// ---------------------------------------------------------------------------
// fp32 -> bf16 conversion (RNE via compiler fptrunc)
// ---------------------------------------------------------------------------
__global__ void cvt_f32_bf16_kernel(const float* __restrict__ s,
                                    bf16* __restrict__ d, int n) {
  int i = blockIdx.x * blockDim.x + threadIdx.x;
  int stride = gridDim.x * blockDim.x;
  for (; i < n; i += stride) d[i] = (bf16)s[i];
}

// Build a 16x32 bf16 A-fragment for one lane from an LDS row (row-major, 32 cols).
// Lane m = lane&15 holds row m; lanes 0-15 use K-base 0, lanes 16-31 use K-base 16.
__device__ __forceinline__ v16bf load_a_frag_lds(const bf16* tile_row, int akb) {
  const v8bf* p = (const v8bf*)tile_row;        // 4 chunks of 8 bf16 (16B each)
  v8bf lo = p[akb >> 2];                        // cols [2*akb, 2*akb+8)
  v8bf hi = p[(akb >> 2) + 1];                  // cols [2*akb+8, 2*akb+16)
  return __builtin_shufflevector(lo, hi, 0, 1, 2, 3, 4, 5, 6, 7,
                                         8, 9, 10, 11, 12, 13, 14, 15);
}

// ---------------------------------------------------------------------------
// Accumulate one GEMM source:  acc += A(64xK tile rows rb..) @ W^T cols gcol..
// K, LDA, LDW are compile-time, so for K=512 all staging / fragment guards
// are provably true and vanish; only the K=16 x-input keeps a (static) pad.
// ---------------------------------------------------------------------------
template <int K, int LDA, int LDW>
__device__ __forceinline__ void gemm_src(
    const bf16* __restrict__ A, const bf16* __restrict__ W,
    int rb, int srow, int schunk, v8bf* sdst,
    int gcol, int nlane, int kb, int m, int akb,
    bf16* Atile, v8f (&acc)[4])
{
  const bf16* const abase = A + (size_t)(rb + srow) * LDA + schunk;
  const bf16* const wbase = W + (size_t)(gcol + nlane) * LDW + kb;
  constexpr int Kp = (K + 31) & ~31;

  #pragma unroll 1
  for (int k0 = 0; k0 < Kp; k0 += 32) {
    __syncthreads();                       // protect Atile reuse
    {
      v8bf av = {};
      if (k0 + schunk + 8 <= K) av = *(const v8bf*)(abase + k0);
      *sdst = av;                          // one ds_store_b128 per thread
    }
    __syncthreads();

    // B fragment: W is (4H,K) row-major -> B[k,n] = W[n,k], contiguous in k
    v16bf bfrag = {};
    if (k0 + kb + 16 <= K) {
      bfrag = *(const v16bf*)(wbase + k0); // 32B contiguous, 2x b128 loads
      if constexpr (K >= 64)
        if (k0 + 32 < K) __builtin_prefetch(wbase + k0 + 32, 0, 3);
    }

    #pragma unroll
    for (int mt = 0; mt < 4; ++mt) {
      v16bf afrag = load_a_frag_lds(&Atile[(mt * 16 + m) * 32], akb);
      acc[mt] = __builtin_amdgcn_wmma_f32_16x16x32_bf16(
          false, afrag, false, bfrag, (short)0, acc[mt], false, false);
    }
  }
}

// ---------------------------------------------------------------------------
// One LSTM timestep (one layer), fused GEMM + gate activations + cell update.
//   gates(B,4H) = A0 @ W0^T [+ A1 @ W1^T] (+ bias) (+ Padd)
//   c = sigm(f)*c + sigm(i)*tanh(g);  h = sigm(o)*tanh(c)
// Grid (16,2), block 256 (8 waves). WG = 64 rows x (32 cols of each gate
// quadrant). Wave w -> quadrant q=w>>1, half hf=w&1; 4 M-tiles of 16 rows.
// RAW=true  -> write raw GEMM result (decoder input projection) and exit.
// HASP=true -> add precomputed per-row B x 4H term (decoder layer 0).
// ---------------------------------------------------------------------------
template <int K0, int LDA0, int LDW0, int K1, int LDA1, int LDW1,
          bool RAW, bool HASP>
__global__ __launch_bounds__(256) void lstm_step_kernel(
    const bf16* __restrict__ A0, const bf16* __restrict__ W0,
    const bf16* __restrict__ A1, const bf16* __restrict__ W1,
    const float* __restrict__ bias,     // 4H
    const float* __restrict__ Padd,     // B x 4H (HASP)
    float* __restrict__ c_state,        // B x H (in/out)
    float* __restrict__ h_state,        // B x H (out, fp32)
    bf16*  __restrict__ h_bf_out,       // B x H (out, bf16, ping-pong)
    float* __restrict__ rawOut)         // raw B x 4H output (RAW)
{
  __shared__ bf16  Atile[64 * 32];      // 4 KB activation tile
  __shared__ float gates[64 * 128];     // 32 KB: 64 rows x (4 quadrants x 32)

  const int n0    = blockIdx.x * 32;    // col base inside each gate quadrant
  const int rb    = blockIdx.y * 64;    // row (batch) base
  const int w     = threadIdx.x >> 5;
  const int lane  = threadIdx.x & 31;
  const int q     = w >> 1;
  const int hf    = w & 1;
  const int gcol  = q * H_ + n0 + hf * 16;   // gate column base (0..2047)
  const int nlane = lane & 15;
  const int kb    = (lane < 16) ? 0 : 16;    // B-fragment K base
  const int akb   = (lane < 16) ? 0 : 8;     // A-fragment K base
  const int m     = lane & 15;               // A row within tile
  const int ro    = (lane < 16) ? 0 : 8;     // C/D row offset for this lane

  // staging assignment: thread -> (row, 8-col chunk); 256 thr x 8 = 64x32
  const int srow   = threadIdx.x >> 2;       // 0..63
  const int schunk = (threadIdx.x & 3) * 8;  // 0,8,16,24
  v8bf* const sdst = (v8bf*)&Atile[srow * 32 + schunk];

  v8f acc[4] = {};

  if constexpr (K0 > 0)
    gemm_src<K0, LDA0, LDW0>(A0, W0, rb, srow, schunk, sdst,
                             gcol, nlane, kb, m, akb, Atile, acc);
  if constexpr (K1 > 0)
    gemm_src<K1, LDA1, LDW1>(A1, W1, rb, srow, schunk, sdst,
                             gcol, nlane, kb, m, akb, Atile, acc);

  if constexpr (RAW) {  // raw GEMM mode (decoder input projection)
    #pragma unroll
    for (int mt = 0; mt < 4; ++mt)
      #pragma unroll
      for (int r = 0; r < 8; ++r)
        rawOut[(size_t)(rb + mt * 16 + r + ro) * G4H + gcol + nlane] = acc[mt][r];
    return;
  } else {
    (void)rawOut;
    // spread accumulators to LDS so each thread can gather its i,f,g,o set
    __syncthreads();
    {
      const int gc = q * 32 + hf * 16 + nlane;     // 0..127
      #pragma unroll
      for (int mt = 0; mt < 4; ++mt)
        #pragma unroll
        for (int r = 0; r < 8; ++r)
          gates[(mt * 16 + r + ro) * 128 + gc] = acc[mt][r];
    }
    __syncthreads();

    // fused LSTM cell update: 64 rows x 32 hidden cols, 8 elems/thread
    for (int e = threadIdx.x; e < 64 * 32; e += 256) {
      int row = e >> 5, n = e & 31;
      int gj  = n0 + n;                 // hidden index 0..511
      int br  = rb + row;               // batch index
      float gi = gates[row * 128 +      n] + bias[         gj];
      float gf = gates[row * 128 + 32 + n] + bias[H_     + gj];
      float gg = gates[row * 128 + 64 + n] + bias[2 * H_ + gj];
      float go = gates[row * 128 + 96 + n] + bias[3 * H_ + gj];
      if constexpr (HASP) {
        const float* p = Padd + (size_t)br * G4H;
        gi += p[gj]; gf += p[H_ + gj]; gg += p[2 * H_ + gj]; go += p[3 * H_ + gj];
      } else {
        (void)Padd;
      }
      gi = 1.0f / (1.0f + __expf(-gi));
      gf = 1.0f / (1.0f + __expf(-gf));
      go = 1.0f / (1.0f + __expf(-go));
      gg = tanhf(gg);
      size_t idx = (size_t)br * H_ + gj;
      float c = gf * c_state[idx] + gi * gg;
      float h = go * tanhf(c);
      c_state[idx]  = c;
      h_state[idx]  = h;
      h_bf_out[idx] = (bf16)h;
    }
  }
}

// ---------------------------------------------------------------------------
// Output projection for one timestep: out[:, t, :] = h(128x512) @ Wout^T + b.
// Single WG, 8 waves, each wave owns a 16x16 tile (N=16 fits one WMMA tile).
// ---------------------------------------------------------------------------
__global__ __launch_bounds__(256) void out_proj_kernel(
    const bf16* __restrict__ A,    // 128 x 512 bf16 (decoder layer-1 h)
    const bf16* __restrict__ Wo,   // 16 x 512 bf16
    const float* __restrict__ bo,  // 16
    float* __restrict__ out, int t)
{
  const int w     = threadIdx.x >> 5;
  const int lane  = threadIdx.x & 31;
  const int m0    = w * 16;
  const int nlane = lane & 15;
  const int kb    = (lane < 16) ? 0 : 16;
  const int akb   = (lane < 16) ? 0 : 8;
  const int m     = lane & 15;
  const int ro    = (lane < 16) ? 0 : 8;

  const bf16* const wbase = Wo + (size_t)nlane * H_ + kb;
  const bf16* const abase = A + (size_t)(m0 + m) * H_;

  v8f acc = {};
  #pragma unroll 1
  for (int k0 = 0; k0 < H_; k0 += 32) {
    v16bf bfrag = *(const v16bf*)(wbase + k0);
    const v8bf* ap = (const v8bf*)(abase + k0);
    v8bf lo = ap[akb >> 2], hi = ap[(akb >> 2) + 1];
    v16bf afrag = __builtin_shufflevector(lo, hi, 0, 1, 2, 3, 4, 5, 6, 7,
                                                  8, 9, 10, 11, 12, 13, 14, 15);
    acc = __builtin_amdgcn_wmma_f32_16x16x32_bf16(
        false, afrag, false, bfrag, (short)0, acc, false, false);
  }
  #pragma unroll
  for (int r = 0; r < 8; ++r) {
    int row = m0 + r + ro;
    out[((size_t)row * T_ + t) * D_ + nlane] = acc[r] + bo[nlane];
  }
}

// ---------------------------------------------------------------------------
extern "C" void kernel_launch(void* const* d_in, const int* in_sizes, int n_in,
                              void* d_out, int out_size, void* d_ws, size_t ws_size,
                              hipStream_t stream) {
  (void)in_sizes; (void)n_in; (void)out_size; (void)ws_size;
  const float* x     = (const float*)d_in[0];
  const float* eWih0 = (const float*)d_in[1];
  const float* eWhh0 = (const float*)d_in[2];
  const float* eb0   = (const float*)d_in[3];
  const float* eWih1 = (const float*)d_in[4];
  const float* eWhh1 = (const float*)d_in[5];
  const float* eb1   = (const float*)d_in[6];
  const float* dWih0 = (const float*)d_in[7];
  const float* dWhh0 = (const float*)d_in[8];
  const float* db0   = (const float*)d_in[9];
  const float* dWih1 = (const float*)d_in[10];
  const float* dWhh1 = (const float*)d_in[11];
  const float* db1   = (const float*)d_in[12];
  const float* Wout  = (const float*)d_in[13];
  const float* bout  = (const float*)d_in[14];
  float* out = (float*)d_out;

  char* ws = (char*)d_ws;
  size_t off = 0;
  auto alloc = [&](size_t bytes) -> char* {
    off = (off + 255) & ~(size_t)255;
    char* p = ws + off;
    off += bytes;
    return p;
  };

  bf16* xbf    = (bf16*)alloc((size_t)B_ * T_ * D_ * 2);
  bf16* weWih0 = (bf16*)alloc((size_t)G4H * D_ * 2);
  bf16* weWhh0 = (bf16*)alloc((size_t)G4H * H_ * 2);
  bf16* weWih1 = (bf16*)alloc((size_t)G4H * H_ * 2);
  bf16* weWhh1 = (bf16*)alloc((size_t)G4H * H_ * 2);
  bf16* wdWih0 = (bf16*)alloc((size_t)G4H * H_ * 2);
  bf16* wdWhh0 = (bf16*)alloc((size_t)G4H * H_ * 2);
  bf16* wdWih1 = (bf16*)alloc((size_t)G4H * H_ * 2);
  bf16* wdWhh1 = (bf16*)alloc((size_t)G4H * H_ * 2);
  bf16* wWout  = (bf16*)alloc((size_t)D_ * H_ * 2);
  bf16* hbf0[2], * hbf1[2];
  hbf0[0] = (bf16*)alloc((size_t)B_ * H_ * 2);
  hbf0[1] = (bf16*)alloc((size_t)B_ * H_ * 2);
  hbf1[0] = (bf16*)alloc((size_t)B_ * H_ * 2);
  hbf1[1] = (bf16*)alloc((size_t)B_ * H_ * 2);
  float* c0s = (float*)alloc((size_t)B_ * H_ * 4);
  float* c1s = (float*)alloc((size_t)B_ * H_ * 4);
  float* h0s = (float*)alloc((size_t)B_ * H_ * 4);
  float* h1s = (float*)alloc((size_t)B_ * H_ * 4);
  float* Pd0 = (float*)alloc((size_t)B_ * G4H * 4);

  auto cvt = [&](const float* s, bf16* d, int n) {
    int g = (n + 255) / 256; if (g > 2048) g = 2048;
    cvt_f32_bf16_kernel<<<dim3(g), dim3(256), 0, stream>>>(s, d, n);
  };
  cvt(x,     xbf,    B_ * T_ * D_);
  cvt(eWih0, weWih0, G4H * D_);
  cvt(eWhh0, weWhh0, G4H * H_);
  cvt(eWih1, weWih1, G4H * H_);
  cvt(eWhh1, weWhh1, G4H * H_);
  cvt(dWih0, wdWih0, G4H * H_);
  cvt(dWhh0, wdWhh0, G4H * H_);
  cvt(dWih1, wdWih1, G4H * H_);
  cvt(dWhh1, wdWhh1, G4H * H_);
  cvt(Wout,  wWout,  D_ * H_);

  // zero initial states (only parity-0 bf16 buffers are read at t=0)
  hipMemsetAsync(hbf0[0], 0, (size_t)B_ * H_ * 2, stream);
  hipMemsetAsync(hbf1[0], 0, (size_t)B_ * H_ * 2, stream);
  hipMemsetAsync(c0s, 0, (size_t)B_ * H_ * 4, stream);
  hipMemsetAsync(c1s, 0, (size_t)B_ * H_ * 4, stream);

  dim3 sgrid(16, 2), sblk(256);

  // ---------------- encoder (2 fused stacked layers) ----------------
  for (int t = 0; t < T_; ++t) {
    int pi = t & 1, po = (t + 1) & 1;
    // layer 0: gates = x_t @ Wih0^T (K=16, zero-padded) + h0 @ Whh0^T
    lstm_step_kernel<D_, T_ * D_, D_, H_, H_, H_, false, false>
        <<<sgrid, sblk, 0, stream>>>(
        xbf + (size_t)t * D_, weWih0,
        hbf0[pi], weWhh0,
        eb0, nullptr, c0s, h0s, hbf0[po], nullptr);
    // layer 1 consumes layer-0 h at the same t (no (B,T,H) intermediate)
    lstm_step_kernel<H_, H_, H_, H_, H_, H_, false, false>
        <<<sgrid, sblk, 0, stream>>>(
        hbf0[po], weWih1,
        hbf1[pi], weWhh1,
        eb1, nullptr, c1s, h1s, hbf1[po], nullptr);
  }

  // encoding = final h of encoder layer 1 -> second output
  hipMemcpyAsync(out + (size_t)B_ * T_ * D_, h1s, (size_t)B_ * H_ * 4,
                 hipMemcpyDeviceToDevice, stream);

  // decoder input projection, computed once: Pd0 = encoding @ dec_Wih0^T
  lstm_step_kernel<H_, H_, H_, 0, 0, 0, true, false>
      <<<sgrid, sblk, 0, stream>>>(
      hbf1[0], wdWih0,
      nullptr, nullptr,
      nullptr, nullptr, nullptr, nullptr, nullptr, Pd0);

  // ---------------- decoder (states continue from encoder finals) ----------
  for (int t = 0; t < T_; ++t) {
    int pi = t & 1, po = (t + 1) & 1;
    // layer 0: gates = Pd0 + h @ dWhh0^T + b
    lstm_step_kernel<0, 0, 0, H_, H_, H_, false, true>
        <<<sgrid, sblk, 0, stream>>>(
        nullptr, nullptr,
        hbf0[pi], wdWhh0,
        db0, Pd0, c0s, h0s, hbf0[po], nullptr);
    // layer 1
    lstm_step_kernel<H_, H_, H_, H_, H_, H_, false, false>
        <<<sgrid, sblk, 0, stream>>>(
        hbf0[po], wdWih1,
        hbf1[pi], wdWhh1,
        db1, nullptr, c1s, h1s, hbf1[po], nullptr);
    // fused per-step output projection into reconstructed (B,T,D)
    out_proj_kernel<<<dim3(1), sblk, 0, stream>>>(hbf1[po], wWout, bout, out, t);
  }
}
